// CDAttention_16243566313815
// MI455X (gfx1250) — matrix operand order repaired and checked by reading, not compiled, not run on hardware
//
#include <hip/hip_runtime.h>
#include <hip/hip_bf16.h>

typedef _Float16 v16h __attribute__((ext_vector_type(16)));
typedef float    v8f  __attribute__((ext_vector_type(8)));

#define WMMA_F16(a, b, c) \
    __builtin_amdgcn_wmma_f32_16x16x32_f16(false, (a), false, (b), (short)0, (c), false, false)

static constexpr int Bsz   = 4;
static constexpr int Cch   = 96;     // channels
static constexpr int Nfull = 4096;   // 64*64
static constexpr int NNs   = 1024;   // 32*32
static constexpr int HEADS = 3;
static constexpr int QT    = 2;      // query tiles per wave in attention
static constexpr int PT    = 4;      // pixel tiles per wave in conv1x1

// ---------------------------------------------------------------------------
// Swizzle a [O x 96] f32 weight matrix into WMMA A-fragment order (f16).
// Element layout: frag[((ot*3 + kc)*32 + lane)*16 + j]
//   row = ot*16 + (lane&15); ksel = (lane>>4)*8
//   col = kc*32 + (j<8 ? ksel+j : 16+ksel+(j-8))
// ---------------------------------------------------------------------------
__global__ void swizzle_w_kernel(const float* __restrict__ w, _Float16* __restrict__ frag,
                                 int otiles) {
    int idx  = blockIdx.x * 256 + threadIdx.x;          // exact grid
    int j    = idx & 15;
    int lane = (idx >> 4) & 31;
    int rest = idx >> 9;
    int kc   = rest % 3;
    int ot   = rest / 3;
    if (ot >= otiles) return;
    int r    = lane & 15;
    int ksel = (lane >> 4) * 8;
    int row  = ot * 16 + r;
    int col  = kc * 32 + ((j < 8) ? (ksel + j) : (16 + ksel + (j - 8)));
    frag[idx] = (_Float16)w[row * Cch + col];
}

// ---------------------------------------------------------------------------
// 1x1-conv GEMM via WMMA: out[o,p] = sum_c w[o,c] * src[b,c,n]
// One wave per (o-chunk of 6 tiles, PT pixel tiles). Weight fragments for the
// whole o-chunk are cached in registers once; per pixel tile only the three
// 32x16 B fragments are gathered, then 18 register-only WMMAs run back to back.
// MODE 0: kv -> k_t [bh,key,32] f16, v_nat [bh,32,key] f16, v_f32 [b,c,n]
// MODE 1: q  -> q_t [bh,query,32] f16
// MODE 2: proj -> out f32 [b,c,n] + bias
// ---------------------------------------------------------------------------
template <int MODE>
__global__ void conv1x1_wmma_kernel(const float* __restrict__ src, int Nsp,
                                    const _Float16* __restrict__ wfrag,
                                    _Float16* __restrict__ k_t, _Float16* __restrict__ v_nat,
                                    float* __restrict__ v_f32, _Float16* __restrict__ q_t,
                                    float* __restrict__ out, const float* __restrict__ bias) {
    int lane  = threadIdx.x;
    int col   = lane & 15;
    int kbase = (lane >> 4) * 8;             // B-fragment K base
    int och   = blockIdx.y;                  // o-chunk of 6 output tiles

    // Cache all 6x3 weight A-fragments for this o-chunk in registers.
    const _Float16* wf = wfrag + (size_t)och * 6 * 3 * 512;
    v16h af[6][3];
#pragma unroll
    for (int ot = 0; ot < 6; ++ot)
#pragma unroll
        for (int kc = 0; kc < 3; ++kc)
            af[ot][kc] = *(const v16h*)(wf + (((size_t)ot * 3 + kc) * 32 + lane) * 16);

    for (int pt = 0; pt < PT; ++pt) {
        int tile = blockIdx.x * PT + pt;
        int p    = tile * 16 + col;          // flattened (b, n)
        int b    = p / Nsp;
        int n    = p % Nsp;

        // Gather the three 32x16 B fragments of src (f32 -> f16).
        v16h bf[3];
#pragma unroll
        for (int kc = 0; kc < 3; ++kc) {
            const float* sp = src + ((size_t)b * Cch + kc * 32 + kbase) * Nsp + n;
#pragma unroll
            for (int j = 0; j < 16; ++j) bf[kc][j] = (_Float16)sp[(size_t)(j < 8 ? j : j + 8) * Nsp];
        }

#pragma unroll
        for (int ot = 0; ot < 6; ++ot) {
            v8f acc = {};
            acc = WMMA_F16(af[ot][0], bf[0], acc);
            acc = WMMA_F16(af[ot][1], bf[1], acc);
            acc = WMMA_F16(af[ot][2], bf[2], acc);

            int rowbase = och * 96 + ot * 16 + (lane >> 4) * 8;
#pragma unroll
            for (int r = 0; r < 8; ++r) {
                int   o   = rowbase + r;
                float val = acc[r];
                if (MODE == 0) {
                    int head  = o >> 6;            // 192 ch -> (head, 64)
                    int inner = o & 63;
                    if (inner < 32) {              // k: store [key, d] (transposed)
                        k_t[(((size_t)(b * HEADS + head)) * Nfull + n) * 32 + inner] =
                            (_Float16)val;
                    } else {                       // v: store [d, key] (natural) + f32 copy
                        int d = inner - 32;
                        v_nat[(((size_t)(b * HEADS + head)) * 32 + d) * Nfull + n] =
                            (_Float16)val;
                        v_f32[((size_t)b * Cch + head * 32 + d) * Nfull + n] = val;
                    }
                } else if (MODE == 1) {            // q: store [query, d] (transposed)
                    int head = o >> 5;
                    int d    = o & 31;
                    q_t[(((size_t)(b * HEADS + head)) * NNs + n) * 32 + d] = (_Float16)val;
                } else {                           // proj: f32 + bias
                    out[((size_t)b * Cch + o) * Nfull + n] = val + bias[o];
                }
            }
        }
    }
}

// ---------------------------------------------------------------------------
// 2x2 average pool: x [b,c,64,64] -> x_samp [b,c,32,32]
// ---------------------------------------------------------------------------
__global__ void pool_kernel(const float* __restrict__ x, float* __restrict__ x_samp) {
    int idx = blockIdx.x * 256 + threadIdx.x;  // total Bsz*Cch*NNs
    int m   = idx & (NNs - 1);
    int bc  = idx >> 10;
    int h2  = m >> 5, w2 = m & 31;
    const float* sp = x + (size_t)bc * Nfull + (h2 * 2) * 64 + (w2 * 2);
    x_samp[idx] = 0.25f * (sp[0] + sp[1] + sp[64] + sp[65]);
}

// ---------------------------------------------------------------------------
// Depthwise 5x5 conv (pad 2) + bias on v_f32 -> lepe
// ---------------------------------------------------------------------------
__global__ void lepe_kernel(const float* __restrict__ v_f32, const float* __restrict__ lw,
                            const float* __restrict__ lb, float* __restrict__ lepe) {
    int idx = blockIdx.x * 256 + threadIdx.x;  // total Bsz*Cch*Nfull
    int pix = idx & (Nfull - 1);
    int bc  = idx >> 12;
    int c   = bc % Cch;
    int y   = pix >> 6, xx = pix & 63;
    float s = lb[c];
    const float* wp = lw + c * 25;
    const float* vp = v_f32 + (size_t)bc * Nfull;
#pragma unroll
    for (int dy = 0; dy < 5; ++dy) {
        int yy = y + dy - 2;
        if (yy < 0 || yy >= 64) continue;
#pragma unroll
        for (int dx = 0; dx < 5; ++dx) {
            int xq = xx + dx - 2;
            if (xq < 0 || xq >= 64) continue;
            s += wp[dy * 5 + dx] * vp[yy * 64 + xq];
        }
    }
    lepe[idx] = s;
}

// ---------------------------------------------------------------------------
// Stage-1 global attention (flash style, softmax over 4096 keys).
// One wave per (b, h, block of QT*16 queries). Keys processed 32 at a time:
//   S = K_tile(A) x Q(B)  -> [key, query] C fragments (softmax in-lane + xor16)
//   O += V^T(A) x P(B)    -> [d, query]
// ---------------------------------------------------------------------------
__global__ void attn_kernel(const _Float16* __restrict__ k_t, const _Float16* __restrict__ v_nat,
                            const _Float16* __restrict__ q_t, float* __restrict__ dist) {
    const float scale = 0.17677669529663689f;  // 32^-0.5
    int lane  = threadIdx.x;
    int col   = lane & 15;
    int kb16  = (lane >> 4) * 16;              // B-fragment K base
    int ksel8 = (lane >> 4) * 8;               // unused (A fragments also use kb16)
    int blk   = blockIdx.x;                    // Bsz*HEADS*(NNs/(16*QT)) blocks
    int qblk  = blk % (NNs / (16 * QT));
    int bh    = blk / (NNs / (16 * QT));
    int b     = bh / HEADS, h = bh % HEADS;
    int qbase = qblk * 16 * QT;

    const _Float16* kp = k_t + (size_t)bh * Nfull * 32;   // [key, d]
    const _Float16* vp = v_nat + (size_t)bh * 32 * Nfull; // [d, key]
    const _Float16* qp = q_t + (size_t)bh * NNs * 32;     // [query, d]

    // Q fragments (B layout): contiguous 16 halves per lane.
    v16h qf[QT];
#pragma unroll
    for (int t = 0; t < QT; ++t) {
        const _Float16* pq = qp + (size_t)(qbase + t * 16 + col) * 32 + kb16;
#pragma unroll
        for (int j = 0; j < 16; ++j) qf[t][j] = pq[j];
    }

    v8f   O[QT][2];
    float mrun[QT], lrun[QT];
#pragma unroll
    for (int t = 0; t < QT; ++t) {
        v8f z = {};
        O[t][0] = z; O[t][1] = z;
        mrun[t] = -1e30f; lrun[t] = 0.f;
    }

    for (int kb = 0; kb < Nfull; kb += 32) {
        // A fragments of K (rows = keys), contiguous in d.
        v16h ak0, ak1;
        {
            const _Float16* pk = kp + (size_t)(kb + col) * 32;
#pragma unroll
            for (int j = 0; j < 8; ++j) { ak0[j] = pk[kb16 + j]; ak0[8 + j] = pk[kb16 + 8 + j]; }
            pk += 16 * 32;
#pragma unroll
            for (int j = 0; j < 8; ++j) { ak1[j] = pk[kb16 + j]; ak1[8 + j] = pk[kb16 + 8 + j]; }
        }
        // A fragments of V^T (rows = d), contiguous in key.
        v16h av0, av1;
        {
            const _Float16* pv = vp + (size_t)col * Nfull + kb;
#pragma unroll
            for (int j = 0; j < 8; ++j) { av0[j] = pv[kb16 + j]; av0[8 + j] = pv[kb16 + 8 + j]; }
            pv += (size_t)16 * Nfull;
#pragma unroll
            for (int j = 0; j < 8; ++j) { av1[j] = pv[kb16 + j]; av1[8 + j] = pv[kb16 + 8 + j]; }
        }

#pragma unroll
        for (int t = 0; t < QT; ++t) {
            v8f s0 = {}, s1 = {};
            s0 = WMMA_F16(ak0, qf[t], s0);   // keys kb..kb+15
            s1 = WMMA_F16(ak1, qf[t], s1);   // keys kb+16..kb+31

            float mloc = -1e30f;
#pragma unroll
            for (int r = 0; r < 8; ++r) {
                float a = s0[r] * scale; s0[r] = a; mloc = fmaxf(mloc, a);
                float c2 = s1[r] * scale; s1[r] = c2; mloc = fmaxf(mloc, c2);
            }
            mloc = fmaxf(mloc, __shfl_xor(mloc, 16, 32));
            float mnew  = fmaxf(mrun[t], mloc);
            float alpha = __expf(mrun[t] - mnew);
            mrun[t] = mnew;

            float lsum = 0.f;
#pragma unroll
            for (int r = 0; r < 8; ++r) {
                s0[r] = __expf(s0[r] - mnew);
                s1[r] = __expf(s1[r] - mnew);
                lsum += s0[r] + s1[r];
            }
            lsum += __shfl_xor(lsum, 16, 32);
            lrun[t] = lrun[t] * alpha + lsum;

#pragma unroll
            for (int r = 0; r < 8; ++r) { O[t][0][r] *= alpha; O[t][1][r] *= alpha; }

            // Repack P (C layout [key,query]) into PV B fragment (K=32 keys).
            v16h pf;
#pragma unroll
            for (int r = 0; r < 8; ++r) {
                float x0 = __shfl_xor(s0[r], 16, 32);
                float x1 = __shfl_xor(s1[r], 16, 32);
                pf[r]     = (lane < 16) ? (_Float16)s0[r] : (_Float16)x1;
                pf[8 + r] = (lane < 16) ? (_Float16)x0    : (_Float16)s1[r];
            }
            O[t][0] = WMMA_F16(av0, pf, O[t][0]);  // d 0..15
            O[t][1] = WMMA_F16(av1, pf, O[t][1]);  // d 16..31
        }
    }

    // Write distribution [b, c=h*32+d, m] in f32.
#pragma unroll
    for (int t = 0; t < QT; ++t) {
        float inv = 1.f / lrun[t];
        int   m   = qbase + t * 16 + col;
#pragma unroll
        for (int dt = 0; dt < 2; ++dt) {
#pragma unroll
            for (int r = 0; r < 8; ++r) {
                int d = dt * 16 + (lane >> 4) * 8 + r;
                dist[((size_t)b * Cch + h * 32 + d) * NNs + m] = O[t][dt][r] * inv;
            }
        }
    }
}

// ---------------------------------------------------------------------------
// Stage-2 local 9-neighbor attention + lepe add. One thread per (b, n2, p).
// ---------------------------------------------------------------------------
__global__ void stage2_kernel(const float* __restrict__ x, const float* __restrict__ x_samp,
                              const float* __restrict__ dist, const float* __restrict__ lepe,
                              float* __restrict__ tmp) {
    const float dsc = 0.10206207261596577f;    // 96^-0.5
    int idx = blockIdx.x * 256 + threadIdx.x;  // total Bsz*NNs*4
    int p   = idx & 3;
    int n2  = (idx >> 2) & (NNs - 1);
    int b   = idx >> 12;
    int h2  = n2 >> 5, w2 = n2 & 31;
    int r1  = p >> 1, r2 = p & 1;
    int pix = (h2 * 2 + r1) * 64 + (w2 * 2 + r2);

    const float* xb = x + (size_t)b * Cch * Nfull;
    const float* sb = x_samp + (size_t)b * Cch * NNs;
    const float* db = dist + (size_t)b * Cch * NNs;

    float xr[Cch];
#pragma unroll
    for (int c = 0; c < Cch; ++c) xr[c] = xb[(size_t)c * Nfull + pix];

    int nbr[9];
#pragma unroll
    for (int k = 0; k < 9; ++k) {
        int hn = h2 + k / 3 - 1, wn = w2 + k % 3 - 1;
        nbr[k] = (hn >= 0 && hn < 32 && wn >= 0 && wn < 32) ? hn * 32 + wn : -1;
    }

    float w9[9];
    float mx = -1e30f;
#pragma unroll
    for (int k = 0; k < 9; ++k) {
        float s = 0.f;
        if (nbr[k] >= 0) {
            const float* sp = sb + nbr[k];
#pragma unroll
            for (int c = 0; c < Cch; ++c) s += xr[c] * sp[(size_t)c * NNs];
        }
        s *= dsc;
        w9[k] = s;
        mx = fmaxf(mx, s);
    }
    float se = 0.f;
#pragma unroll
    for (int k = 0; k < 9; ++k) { w9[k] = __expf(w9[k] - mx); se += w9[k]; }
    float inv = 1.f / se;
#pragma unroll
    for (int k = 0; k < 9; ++k) w9[k] *= inv;

#pragma unroll 4
    for (int c = 0; c < Cch; ++c) {
        float f = 0.f;
        const float* dp = db + (size_t)c * NNs;
#pragma unroll
        for (int k = 0; k < 9; ++k)
            if (nbr[k] >= 0) f += w9[k] * dp[nbr[k]];
        size_t oidx = ((size_t)b * Cch + c) * Nfull + pix;
        tmp[oidx] = f + lepe[oidx];
    }
}

// ---------------------------------------------------------------------------
extern "C" void kernel_launch(void* const* d_in, const int* in_sizes, int n_in,
                              void* d_out, int out_size, void* d_ws, size_t ws_size,
                              hipStream_t stream) {
    const float* x      = (const float*)d_in[0];
    const float* kv_w   = (const float*)d_in[1];
    const float* q_w    = (const float*)d_in[2];
    const float* lepe_w = (const float*)d_in[3];
    const float* lepe_b = (const float*)d_in[4];
    const float* proj_w = (const float*)d_in[5];
    const float* proj_b = (const float*)d_in[6];
    float* out = (float*)d_out;
    (void)in_sizes; (void)n_in; (void)out_size; (void)ws_size;

    char* ws = (char*)d_ws;
    size_t off = 0;
    auto carve = [&](size_t bytes) -> char* {
        char* pp = ws + off;
        off += (bytes + 255) & ~(size_t)255;
        return pp;
    };
    _Float16* wfrag_kv = (_Float16*)carve((size_t)12 * 3 * 512 * 2);
    _Float16* wfrag_q  = (_Float16*)carve((size_t)6 * 3 * 512 * 2);
    _Float16* wfrag_p  = (_Float16*)carve((size_t)6 * 3 * 512 * 2);
    _Float16* k_t      = (_Float16*)carve((size_t)Bsz * HEADS * Nfull * 32 * 2);
    _Float16* v_nat    = (_Float16*)carve((size_t)Bsz * HEADS * 32 * Nfull * 2);
    _Float16* q_t      = (_Float16*)carve((size_t)Bsz * HEADS * NNs * 32 * 2);
    float*    v_f32    = (float*)carve((size_t)Bsz * Cch * Nfull * 4);
    float*    x_samp   = (float*)carve((size_t)Bsz * Cch * NNs * 4);
    float*    dist     = (float*)carve((size_t)Bsz * Cch * NNs * 4);
    float*    lepe     = (float*)carve((size_t)Bsz * Cch * Nfull * 4);
    float*    tmp      = (float*)carve((size_t)Bsz * Cch * Nfull * 4);

    // Weight pre-swizzle into A-fragment order.
    swizzle_w_kernel<<<72, 256, 0, stream>>>(kv_w, wfrag_kv, 12);
    swizzle_w_kernel<<<36, 256, 0, stream>>>(q_w, wfrag_q, 6);
    swizzle_w_kernel<<<36, 256, 0, stream>>>(proj_w, wfrag_p, 6);

    // kv = kv_w @ x -> k_t (f16, [key,d]), v_nat (f16, [d,key]), v_f32
    {
        dim3 grid((Bsz * Nfull / 16) / PT, 2);   // 2 o-chunks of 6 tiles (192 ch)
        conv1x1_wmma_kernel<0><<<grid, 32, 0, stream>>>(
            x, Nfull, wfrag_kv, k_t, v_nat, v_f32, nullptr, nullptr, nullptr);
    }

    // 2x2 average pool
    pool_kernel<<<(Bsz * Cch * NNs) / 256, 256, 0, stream>>>(x, x_samp);

    // q = q_w @ x_samp -> q_t (f16, [query,d])
    {
        dim3 grid((Bsz * NNs / 16) / PT, 1);
        conv1x1_wmma_kernel<1><<<grid, 32, 0, stream>>>(
            x_samp, NNs, wfrag_q, nullptr, nullptr, nullptr, q_t, nullptr, nullptr);
    }

    // lepe depthwise 5x5 + bias on v_f32
    lepe_kernel<<<(Bsz * Cch * Nfull) / 256, 256, 0, stream>>>(v_f32, lepe_w, lepe_b, lepe);

    // Stage-1 flash attention -> dist [b,c,m]
    attn_kernel<<<Bsz * HEADS * (NNs / (16 * QT)), 32, 0, stream>>>(k_t, v_nat, q_t, dist);

    // Stage-2 local attention + lepe -> tmp [b,c,pix]
    stage2_kernel<<<(Bsz * NNs * 4) / 256, 256, 0, stream>>>(x, x_samp, dist, lepe, tmp);

    // proj = proj_w @ tmp + proj_b -> out
    {
        dim3 grid((Bsz * Nfull / 16) / PT, 1);
        conv1x1_wmma_kernel<2><<<grid, 32, 0, stream>>>(
            tmp, Nfull, wfrag_p, nullptr, nullptr, nullptr, nullptr, out, proj_b);
    }
}
